// Attention_67946382623269
// MI455X (gfx1250) — matrix operand order repaired and checked by reading, not compiled, not run on hardware
//
#include <hip/hip_runtime.h>

typedef __bf16 bf16;
typedef __attribute__((ext_vector_type(16))) __bf16 v16bf;
typedef __attribute__((ext_vector_type(8)))  float  v8f;

#define L_SEQ   2048
#define D_MODEL 2048
#define NH      32
#define NKV     8
#define HDIM    64

#define WMMA_BF16(A,B,C) \
  __builtin_amdgcn_wmma_f32_16x16x32_bf16(false,(A),false,(B),(short)0,(C),false,false)

// ---------------------------------------------------------------- pre-pass ---
__global__ void f2bf_kernel(const float* __restrict__ src, bf16* __restrict__ dst, int n) {
  int i = blockIdx.x * blockDim.x + threadIdx.x;
  if (i < n) dst[i] = (bf16)src[i];
}

// w: fp32 [rows(K)][cols(N)] row-major  ->  wt: bf16 [N][K]
__global__ void transpose_w_kernel(const float* __restrict__ w, bf16* __restrict__ wt,
                                   int rows, int cols) {
  int i = blockIdx.x * blockDim.x + threadIdx.x;
  if (i >= rows * cols) return;
  int n = i / rows, k = i % rows;
  wt[i] = (bf16)w[(size_t)k * cols + n];
}

// In-place RoPE on bf16 [L][nH*64]; one thread per (pos, head, pair j)
__global__ void rope_kernel(bf16* buf, const float* __restrict__ cosp,
                            const float* __restrict__ sinp, int nH, int n) {
  int i = blockIdx.x * blockDim.x + threadIdx.x;
  if (i >= n) return;
  int per = nH * 32;
  int pos = i / per;
  int r   = i % per;
  int h   = r >> 5;
  int j   = r & 31;
  size_t base = (size_t)pos * nH * 64 + h * 64 + 2 * j;
  float xr = (float)buf[base];
  float xi = (float)buf[base + 1];
  float c = cosp[pos * 32 + j];
  float s = sinp[pos * 32 + j];
  buf[base]     = (bf16)(xr * c - xi * s);
  buf[base + 1] = (bf16)(xr * s + xi * c);
}

// v0: bf16 [L][NKV*64] -> vt: bf16 [NKV*64][L]
__global__ void vtrans_kernel(const bf16* __restrict__ v0, bf16* __restrict__ vt, int n) {
  int i = blockIdx.x * blockDim.x + threadIdx.x;
  if (i >= n) return;
  int c  = i >> 11;      // kv*64 + d
  int kk = i & 2047;     // key position
  vt[i] = v0[(size_t)kk * (NKV * HDIM) + c];
}

// ------------------------------------------------------------------- GEMM ---
// C[M][N] = A[M][K] * Bt[N][K]^T ; one wave per 64x64 output tile.
template <typename OutT>
__global__ void __launch_bounds__(32)
gemm_kernel(const bf16* __restrict__ A, const bf16* __restrict__ Bt,
            OutT* __restrict__ C, int lda, int ldb, int ldc, int K) {
  const int lane  = threadIdx.x & 31;
  const int half  = lane >> 4;
  const int ln15  = lane & 15;
  const int m0    = blockIdx.y * 64;
  const int n0    = blockIdx.x * 64;
  const int abase = half * 16;   // A frag K base (contiguous 16)
  const int bbase = half * 16;   // B frag K base (contiguous 16)

  v8f acc[16];
#pragma unroll
  for (int i = 0; i < 16; i++) acc[i] = v8f{};

  for (int k0 = 0; k0 < K; k0 += 32) {
    v16bf a[4], b[4];
#pragma unroll
    for (int mt = 0; mt < 4; mt++) {
      const uint4* p = reinterpret_cast<const uint4*>(
          A + (size_t)(m0 + mt * 16 + ln15) * lda + k0 + abase);
      union { uint4 u[2]; v16bf v; } ua;
      ua.u[0] = p[0];
      ua.u[1] = p[1];           // K = base+8..base+15
      a[mt] = ua.v;
    }
#pragma unroll
    for (int nt = 0; nt < 4; nt++) {
      const uint4* p = reinterpret_cast<const uint4*>(
          Bt + (size_t)(n0 + nt * 16 + ln15) * ldb + k0 + bbase);
      union { uint4 u[2]; v16bf v; } ub;
      ub.u[0] = p[0];
      ub.u[1] = p[1];           // 16 contiguous K values
      b[nt] = ub.v;
    }
    if (k0 + 32 < K) {          // hint next A tile into L2 (global_prefetch_b8)
      __builtin_prefetch(A + (size_t)(m0 + ln15) * lda + k0 + 32, 0, 1);
    }
#pragma unroll
    for (int mt = 0; mt < 4; mt++)
#pragma unroll
      for (int nt = 0; nt < 4; nt++)
        acc[mt * 4 + nt] = WMMA_BF16(a[mt], b[nt], acc[mt * 4 + nt]);
  }

#pragma unroll
  for (int mt = 0; mt < 4; mt++)
#pragma unroll
    for (int nt = 0; nt < 4; nt++)
#pragma unroll
      for (int r = 0; r < 8; r++) {
        int row = m0 + mt * 16 + r + half * 8;
        int col = n0 + nt * 16 + ln15;
        C[(size_t)row * ldc + col] = (OutT)acc[mt * 4 + nt][r];
      }
}

// -------------------------------------------------------- flash attention ---
// One wave per (head, 16-query tile); streams 32-key chunks with online softmax.
// q: bf16 [L][NH*64] (roped), k: bf16 [L][NKV*64] (roped), vt: bf16 [NKV*64][L]
// o_out: bf16 [L][NH*64]
__global__ void __launch_bounds__(128)
attn_kernel(const bf16* __restrict__ q, const bf16* __restrict__ k,
            const bf16* __restrict__ vt, bf16* __restrict__ o_out) {
  __shared__ alignas(16) bf16 pbuf[4][16 * 32];

  const int wave = threadIdx.x >> 5;
  const int lane = threadIdx.x & 31;
  const int half = lane >> 4;
  const int ln15 = lane & 15;
  const int gw   = blockIdx.x * 4 + wave;
  const int head = gw >> 7;          // 32 heads
  const int tile = gw & 127;         // 128 query tiles
  const int kvh  = head >> 2;        // GQA: 4 heads per kv head
  const int q0   = tile * 16;
  const int abase = half * 16;
  const int bbase = half * 16;

  // Q fragments: rows q0..q0+15, d-chunks {0..31},{32..63}
  v16bf a0, a1;
  {
    const bf16* qrow = q + (size_t)(q0 + ln15) * D_MODEL + head * HDIM;
    union { uint4 u[2]; v16bf v; } ua;
    const uint4* p0 = reinterpret_cast<const uint4*>(qrow + abase);
    ua.u[0] = p0[0]; ua.u[1] = p0[1]; a0 = ua.v;
    const uint4* p1 = reinterpret_cast<const uint4*>(qrow + 32 + abase);
    ua.u[0] = p1[0]; ua.u[1] = p1[1]; a1 = ua.v;
  }

  v8f oacc[4];
#pragma unroll
  for (int j = 0; j < 4; j++) oacc[j] = v8f{};
  float mrow[8], lrow[8];
#pragma unroll
  for (int r = 0; r < 8; r++) { mrow[r] = -3.0e38f; lrow[r] = 0.0f; }

  bf16* pw = pbuf[wave];

  for (int kk0 = 0; kk0 < q0 + 16; kk0 += 32) {
    // --- scores: S[16q x 32k] = Q[16x64] @ K^T[64x32], two 16-key tiles ---
    v16bf bk[2][2];
#pragma unroll
    for (int t = 0; t < 2; t++) {
      const bf16* krow = k + (size_t)(kk0 + t * 16 + ln15) * (NKV * HDIM) + kvh * HDIM;
#pragma unroll
      for (int c = 0; c < 2; c++) {
        union { uint4 u[2]; v16bf v; } ub;
        const uint4* p = reinterpret_cast<const uint4*>(krow + c * 32 + bbase);
        ub.u[0] = p[0]; ub.u[1] = p[1];
        bk[t][c] = ub.v;
      }
    }
    v8f s0 = v8f{}, s1 = v8f{};
    s0 = WMMA_BF16(a0, bk[0][0], s0);
    s0 = WMMA_BF16(a1, bk[0][1], s0);
    s1 = WMMA_BF16(a0, bk[1][0], s1);
    s1 = WMMA_BF16(a1, bk[1][1], s1);

    // --- scale + causal mask + online softmax (per row, 16-lane groups) ---
#pragma unroll
    for (int r = 0; r < 8; r++) {
      int   qi   = q0 + r + half * 8;
      int   key0 = kk0 + ln15;
      int   key1 = kk0 + 16 + ln15;
      float v0f  = s0[r] * 0.125f + (key0 > qi ? -1.0e9f : 0.0f);
      float v1f  = s1[r] * 0.125f + (key1 > qi ? -1.0e9f : 0.0f);

      float t = fmaxf(v0f, v1f);
      t = fmaxf(t, __shfl_xor(t, 8, 16));
      t = fmaxf(t, __shfl_xor(t, 4, 16));
      t = fmaxf(t, __shfl_xor(t, 2, 16));
      t = fmaxf(t, __shfl_xor(t, 1, 16));

      float newm = fmaxf(mrow[r], t);
      float sc   = __expf(mrow[r] - newm);
      float p0   = __expf(v0f - newm);
      float p1   = __expf(v1f - newm);
      float ps   = p0 + p1;
      ps += __shfl_xor(ps, 8, 16);
      ps += __shfl_xor(ps, 4, 16);
      ps += __shfl_xor(ps, 2, 16);
      ps += __shfl_xor(ps, 1, 16);

      lrow[r] = lrow[r] * sc + ps;
      mrow[r] = newm;
#pragma unroll
      for (int j = 0; j < 4; j++) oacc[j][r] = oacc[j][r] * sc;

      int prow = r + half * 8;
      pw[prow * 32 + ln15]      = (bf16)p0;   // keys kk0..kk0+15
      pw[prow * 32 + 16 + ln15] = (bf16)p1;   // keys kk0+16..kk0+31
    }

    asm volatile("s_wait_dscnt 0" ::: "memory");  // C-layout -> A-layout via LDS

    // --- reload P as WMMA A-fragment (16 rows x 32 keys) ---
    v16bf pf;
    {
      union { uint4 u[2]; v16bf v; } up;
      const uint4* pp = reinterpret_cast<const uint4*>(pw + ln15 * 32 + abase);
      up.u[0] = pp[0]; up.u[1] = pp[1];
      pf = up.v;
    }

    // --- O[16x64] += P[16x32] @ V[32x64] ---
#pragma unroll
    for (int j = 0; j < 4; j++) {
      union { uint4 u[2]; v16bf v; } uv;
      const uint4* pv = reinterpret_cast<const uint4*>(
          vt + (size_t)(kvh * HDIM + j * 16 + ln15) * L_SEQ + kk0 + bbase);
      uv.u[0] = pv[0]; uv.u[1] = pv[1];
      oacc[j] = WMMA_BF16(pf, uv.v, oacc[j]);
    }
  }

  // --- normalize and store bf16 attention output ---
#pragma unroll
  for (int r = 0; r < 8; r++) {
    float inv = 1.0f / lrow[r];
    int   row = q0 + r + half * 8;
#pragma unroll
    for (int j = 0; j < 4; j++) {
      o_out[(size_t)row * D_MODEL + head * HDIM + j * 16 + ln15] =
          (bf16)(oacc[j][r] * inv);
    }
  }
}

// ------------------------------------------------------------------ launch ---
extern "C" void kernel_launch(void* const* d_in, const int* in_sizes, int n_in,
                              void* d_out, int out_size, void* d_ws, size_t ws_size,
                              hipStream_t stream) {
  (void)in_sizes; (void)n_in; (void)out_size; (void)ws_size;
  const float* x  = (const float*)d_in[0];
  const float* wq = (const float*)d_in[1];
  const float* wk = (const float*)d_in[2];
  const float* wv = (const float*)d_in[3];
  const float* wo = (const float*)d_in[4];
  const float* fc = (const float*)d_in[5];
  const float* fs = (const float*)d_in[6];
  // d_in[7] (mask) unused: causal mask applied analytically.

  char*  ws  = (char*)d_ws;
  size_t off = 0;
  auto alloc = [&](size_t elems) -> bf16* {
    bf16* p = (bf16*)(ws + off);
    off += elems * sizeof(bf16);
    off  = (off + 255) & ~(size_t)255;
    return p;
  };
  bf16* xb   = alloc((size_t)L_SEQ * D_MODEL);    // x, bf16
  bf16* wqT  = alloc((size_t)2048 * 2048);        // wq^T
  bf16* wkT  = alloc((size_t)512  * 2048);        // wk^T
  bf16* wvT  = alloc((size_t)512  * 2048);        // wv^T
  bf16* woT  = alloc((size_t)2048 * 2048);        // wo^T
  bf16* qb   = alloc((size_t)2048 * 2048);        // Q proj (roped in place)
  bf16* kb   = alloc((size_t)2048 * 512);         // K proj (roped in place)
  bf16* vb   = alloc((size_t)2048 * 512);         // V proj
  bf16* vt   = alloc((size_t)512  * 2048);        // V transposed per kv head
  bf16* attn = alloc((size_t)2048 * 2048);        // attention output, bf16

  int n;
  n = 2048 * 2048; f2bf_kernel<<<(n + 255) / 256, 256, 0, stream>>>(x, xb, n);
  n = 2048 * 2048; transpose_w_kernel<<<(n + 255) / 256, 256, 0, stream>>>(wq, wqT, 2048, 2048);
  n = 512  * 2048; transpose_w_kernel<<<(n + 255) / 256, 256, 0, stream>>>(wk, wkT, 2048, 512);
  n = 512  * 2048; transpose_w_kernel<<<(n + 255) / 256, 256, 0, stream>>>(wv, wvT, 2048, 512);
  n = 2048 * 2048; transpose_w_kernel<<<(n + 255) / 256, 256, 0, stream>>>(wo, woT, 2048, 2048);

  gemm_kernel<bf16><<<dim3(2048 / 64, 2048 / 64), 32, 0, stream>>>(xb, wqT, qb, 2048, 2048, 2048, 2048);
  gemm_kernel<bf16><<<dim3(512  / 64, 2048 / 64), 32, 0, stream>>>(xb, wkT, kb, 2048, 2048, 512, 2048);
  gemm_kernel<bf16><<<dim3(512  / 64, 2048 / 64), 32, 0, stream>>>(xb, wvT, vb, 2048, 2048, 512, 2048);

  n = 2048 * 32 * 32; rope_kernel<<<(n + 255) / 256, 256, 0, stream>>>(qb, fc, fs, 32, n);
  n = 2048 * 8  * 32; rope_kernel<<<(n + 255) / 256, 256, 0, stream>>>(kb, fc, fs, 8, n);
  n = 512 * 2048;     vtrans_kernel<<<(n + 255) / 256, 256, 0, stream>>>(vb, vt, n);

  attn_kernel<<<1024, 128, 0, stream>>>(qb, kb, vt, attn);

  gemm_kernel<float><<<dim3(2048 / 64, 2048 / 64), 32, 0, stream>>>(
      attn, woT, (float*)d_out, 2048, 2048, 2048, 2048);
}
